// CausalSelfAttention_30511447671668
// MI455X (gfx1250) — compile-verified, matrix operand order checked by reading, not run on hardware
//
#include <hip/hip_runtime.h>
#include <hip/hip_bf16.h>

// ---------------------------------------------------------------------------
// CausalSelfAttention forward for MI455X (gfx1250, wave32, WMMA)
// x[2,2048,1024] @ w_qkv[1024,3072] -> qkv ; flash attention ; @ w_proj
// - v_wmma_f32_16x16x32_bf16 everywhere (f32 accumulate)
// - Weights pre-transposed, V stored [B,H,Dh,T]: all fragments = 2x b128
// - GEMM k-loops software-pipelined (next fragments in flight during WMMA)
// - Attention: 8 waves/block share ping-pong LDS K/V tiles staged with
//   global_load_async_to_lds_b128 (ASYNCcnt overlapped with compute)
// ---------------------------------------------------------------------------

#define BB   2
#define TT   2048
#define DM   1024
#define HH   16
#define DH   64
#define NQKV 3072

typedef __attribute__((ext_vector_type(16))) __bf16 v16bf;
typedef __attribute__((ext_vector_type(8)))  float  v8f;

union Frag16 { v16bf v; unsigned int u[8]; uint4 q2[2]; };
union F32x8  { v8f  v; float f[8]; };

__device__ __forceinline__ unsigned short f2bf(float f) {
  union { __bf16 h; unsigned short u; } cv;
  cv.h = (__bf16)f;
  return cv.u;
}

// A/B-style 16x32 bf16 fragment (ISA 7.12.2): per lane, dwords 0-3 hold
// K = {0..7} + 8*(lane>=16), dwords 4-7 hold K = {16..23} + 8*(lane>=16).
// Both runs are 16 contiguous bytes -> two b128 loads.
__device__ __forceinline__ void load_frag(Frag16& f, const unsigned short* p,
                                          int elemBase, int lane) {
  const int hi = (lane >= 16) ? 8 : 0;
  f.q2[0] = *(const uint4*)(p + elemBase + hi);
  f.q2[1] = *(const uint4*)(p + elemBase + 16 + hi);
}

__device__ __forceinline__ v8f wmma_bf16(v16bf a, v16bf b, v8f c) {
  return __builtin_amdgcn_wmma_f32_16x16x32_bf16(
      false, a, false, b, (short)0, c, false, false);
}

// Async global -> LDS 16-byte copy (per-lane), tracked with ASYNCcnt.
__device__ __forceinline__ void async_copy_b128(unsigned lds_off,
                                                const unsigned short* gaddr) {
  asm volatile("global_load_async_to_lds_b128 %0, %1, off"
               :: "v"(lds_off), "v"((unsigned long long)gaddr)
               : "memory");
}

// ---------------------------------------------------------------------------
// fp32 -> bf16 conversions (straight + transposing for weights)
// ---------------------------------------------------------------------------
__global__ void f32_to_bf16_kernel(const float* __restrict__ src,
                                   unsigned short* __restrict__ dst, int n) {
  int i = blockIdx.x * blockDim.x + threadIdx.x;
  if (i < n) dst[i] = f2bf(src[i]);
}

__global__ void f32_to_bf16_tr_kernel(const float* __restrict__ src,  // [K,N]
                                      unsigned short* __restrict__ dst, // [N,K]
                                      int K, int N) {
  int i = blockIdx.x * blockDim.x + threadIdx.x;
  if (i < K * N) {
    int k = i / N, n = i - k * N;
    dst[n * K + k] = f2bf(src[i]);
  }
}

// ---------------------------------------------------------------------------
// QKV GEMM: [4096,1024]bf16 x Wt[3072,1024]bf16 + bias, pipelined k-loop.
// One wave computes a 16x64 tile (4 subtiles share the A fragment).
// Q,K -> [B,H,T,Dh] bf16 ; V -> [B,H,Dh,T] bf16 (transposed for P@V).
// ---------------------------------------------------------------------------
__global__ void __launch_bounds__(256)
qkv_gemm_kernel(const unsigned short* __restrict__ xb,   // bf16 [4096,1024]
                const unsigned short* __restrict__ wt,   // bf16 [3072,1024]
                const float* __restrict__ bias,
                unsigned short* __restrict__ qb,
                unsigned short* __restrict__ kb,
                unsigned short* __restrict__ vb) {
  const int lane = threadIdx.x & 31;
  const int wave = __builtin_amdgcn_readfirstlane(threadIdx.x >> 5);
  const int NT64 = NQKV / 64;                 // 48
  const int tile = blockIdx.x * 8 + wave;     // 256*48 = 12288 wave-tiles
  const int tm = tile / NT64;
  const int tn64 = tile - tm * NT64;
  const int arow = tm * 16 + (lane & 15);

  F32x8 acc[4];
#pragma unroll
  for (int c = 0; c < 4; ++c)
#pragma unroll
    for (int i = 0; i < 8; ++i) acc[c].f[i] = 0.0f;

  int bcol[4];
#pragma unroll
  for (int c = 0; c < 4; ++c) bcol[c] = (tn64 * 4 + c) * 16 + (lane & 15);

  // Software pipeline: fragments for step k0 live in *_cur while the loads
  // for k0+32 are in flight into *_nxt.
  Frag16 a_cur, b_cur[4];
  load_frag(a_cur, xb, arow * DM, lane);
#pragma unroll
  for (int c = 0; c < 4; ++c) load_frag(b_cur[c], wt, bcol[c] * DM, lane);

  for (int k0 = 0; k0 < DM; k0 += 32) {
    Frag16 a_nxt, b_nxt[4];
    const int kn = k0 + 32;
    if (kn < DM) {
      load_frag(a_nxt, xb, arow * DM + kn, lane);
      __builtin_prefetch(xb + arow * DM + kn + 32, 0, 1);
#pragma unroll
      for (int c = 0; c < 4; ++c) load_frag(b_nxt[c], wt, bcol[c] * DM + kn, lane);
    }
#pragma unroll
    for (int c = 0; c < 4; ++c)
      acc[c].v = wmma_bf16(a_cur.v, b_cur[c].v, acc[c].v);
    a_cur = a_nxt;
#pragma unroll
    for (int c = 0; c < 4; ++c) b_cur[c] = b_nxt[c];
  }

#pragma unroll
  for (int c = 0; c < 4; ++c) {
    const int tn = tn64 * 4 + c;
    const int which = tn >> 6;          // 0=q 1=k 2=v   (scalar)
    const int h = (tn >> 2) & 15;       // head          (scalar)
    const int d = ((tn & 3) * 16) + (lane & 15);
    const float bias_n = bias[tn * 16 + (lane & 15)];
#pragma unroll
    for (int i = 0; i < 8; ++i) {
      const int m = tm * 16 + i + ((lane >= 16) ? 8 : 0);
      const int bidx = m >> 11;
      const int t = m & 2047;
      const unsigned short bits = f2bf(acc[c].f[i] + bias_n);
      if (which == 0)
        qb[((bidx * HH + h) * TT + t) * DH + d] = bits;
      else if (which == 1)
        kb[((bidx * HH + h) * TT + t) * DH + d] = bits;
      else
        vb[((bidx * HH + h) * DH + d) * TT + t] = bits;
    }
  }
}

// ---------------------------------------------------------------------------
// Flash attention, workgroup-cooperative with ping-pong async staging:
//   block = 8 waves = 8 consecutive 16-query tiles of one (b,h).
//   While block i is computed from LDS buffer i&1, block i+1's K (32x64) and
//   Vt (64x32) tiles are async-copied into buffer (i+1)&1 (ASYNCcnt).
// ---------------------------------------------------------------------------
__global__ void __launch_bounds__(256)
attn_kernel(const unsigned short* __restrict__ qbm,  // bf16 [B,H,T,Dh]
            const unsigned short* __restrict__ kbm,  // bf16 [B,H,T,Dh]
            const unsigned short* __restrict__ vtm,  // bf16 [B,H,Dh,T]
            unsigned short* __restrict__ ob) {       // bf16 [B,T,C]
  __shared__ __align__(16) unsigned short kbufA[32 * 64];   // 4 KB
  __shared__ __align__(16) unsigned short kbufB[32 * 64];   // 4 KB
  __shared__ __align__(16) unsigned short vbufA[64 * 32];   // 4 KB
  __shared__ __align__(16) unsigned short vbufB[64 * 32];   // 4 KB
  __shared__ __align__(16) unsigned short plds[8][16 * 32]; // 8 KB

  const int tid = threadIdx.x;
  const int lane = tid & 31;
  const int wave = __builtin_amdgcn_readfirstlane(tid >> 5);
  const int bh = blockIdx.x >> 4;                    // 32 (b,h) pairs
  const int qt = (blockIdx.x & 15) * 8 + wave;       // 128 q-tiles per bh

  const unsigned short* qp = qbm + bh * (TT * DH);
  const unsigned short* kp = kbm + bh * (TT * DH);
  const unsigned short* vt = vtm + bh * (TT * DH);

  Frag16 qf0, qf1;
  const int qrow = qt * 16 + (lane & 15);
  load_frag(qf0, qp, qrow * DH, lane);
  load_frag(qf1, qp, qrow * DH + 32, lane);

  float mi[8], li[8];
  F32x8 o[4];
#pragma unroll
  for (int i = 0; i < 8; ++i) { mi[i] = -3.0e38f; li[i] = 0.0f; }
#pragma unroll
  for (int t = 0; t < 4; ++t)
#pragma unroll
    for (int i = 0; i < 8; ++i) o[t].f[i] = 0.0f;

  // LDS staging coordinates (per thread, 16B each)
  const int krow = tid >> 3, kcol = (tid & 7) * 8;   // K: 32x64
  const int vrow = tid >> 2, vcol = (tid & 3) * 8;   // Vt: 64x32
  const unsigned kldsA = (unsigned)(unsigned long long)(const void*)&kbufA[krow * 64 + kcol];
  const unsigned kldsB = (unsigned)(unsigned long long)(const void*)&kbufB[krow * 64 + kcol];
  const unsigned vldsA = (unsigned)(unsigned long long)(const void*)&vbufA[vrow * 32 + vcol];
  const unsigned vldsB = (unsigned)(unsigned long long)(const void*)&vbufB[vrow * 32 + vcol];

  const int myKend = qt * 16 + 16;                       // this wave's bound
  const int blockKend = ((blockIdx.x & 15) * 8 + 7) * 16 + 16;  // block bound
  const int nb = blockKend >> 5;                         // #32-key blocks

  // Prologue: stage block 0 into buffer A.
  async_copy_b128(kldsA, kp + krow * DH + kcol);
  async_copy_b128(vldsA, vt + vrow * TT + vcol);

  for (int ib = 0; ib < nb; ++ib) {
    const int kb0 = ib * 32;
    if (ib + 1 < nb) {
      // Issue next block's copies into the other buffer, then retire only
      // the current block's pair (async loads complete in order).
      const int kn = kb0 + 32;
      const unsigned kn_lds = ((ib + 1) & 1) ? kldsB : kldsA;
      const unsigned vn_lds = ((ib + 1) & 1) ? vldsB : vldsA;
      async_copy_b128(kn_lds, kp + (kn + krow) * DH + kcol);
      async_copy_b128(vn_lds, vt + vrow * TT + kn + vcol);
      asm volatile("s_wait_asynccnt 0x2" ::: "memory");
    } else {
      asm volatile("s_wait_asynccnt 0x0" ::: "memory");
    }
    __syncthreads();                                    // block ib visible

    const unsigned short* kbuf = (ib & 1) ? kbufB : kbufA;
    const unsigned short* vbuf = (ib & 1) ? vbufB : vbufA;

    if (kb0 < myKend) {   // wave-uniform causal skip (EXEC stays all-ones)
      // ---- S = Q K^T from LDS K tile --------------------------------------
      F32x8 s0, s1;
#pragma unroll
      for (int i = 0; i < 8; ++i) { s0.f[i] = 0.0f; s1.f[i] = 0.0f; }
      const int l0 = (lane & 15);            // local key row (tile 0)
      {
        Frag16 kf;
        load_frag(kf, kbuf, l0 * 64, lane);
        s0.v = wmma_bf16(qf0.v, kf.v, s0.v);
        load_frag(kf, kbuf, l0 * 64 + 32, lane);
        s0.v = wmma_bf16(qf1.v, kf.v, s0.v);
        load_frag(kf, kbuf, (l0 + 16) * 64, lane);
        s1.v = wmma_bf16(qf0.v, kf.v, s1.v);
        load_frag(kf, kbuf, (l0 + 16) * 64 + 32, lane);
        s1.v = wmma_bf16(qf1.v, kf.v, s1.v);
      }
      const int key0 = kb0 + l0;
      const int key1 = key0 + 16;

      // ---- scale + causal mask + row max ----------------------------------
      float rm[8], ls[8], al[8];
#pragma unroll
      for (int i = 0; i < 8; ++i) {
        const int rowq = qt * 16 + i + ((lane >= 16) ? 8 : 0);
        float v0 = s0.f[i] * 0.125f;
        float v1 = s1.f[i] * 0.125f;
        if (key0 > rowq) v0 = -3.0e38f;
        if (key1 > rowq) v1 = -3.0e38f;
        s0.f[i] = v0; s1.f[i] = v1;
        rm[i] = fmaxf(v0, v1);
      }
#pragma unroll
      for (int mask = 1; mask <= 8; mask <<= 1)
#pragma unroll
        for (int i = 0; i < 8; ++i)
          rm[i] = fmaxf(rm[i], __shfl_xor(rm[i], mask, 32));

      // ---- exp, P -> LDS (bf16), partial row sums -------------------------
#pragma unroll
      for (int i = 0; i < 8; ++i) {
        const float mnew = fmaxf(mi[i], rm[i]);
        al[i] = __expf(mi[i] - mnew);
        const float p0 = __expf(s0.f[i] - mnew);
        const float p1 = __expf(s1.f[i] - mnew);
        const int rloc = i + ((lane >= 16) ? 8 : 0);
        plds[wave][rloc * 32 + (lane & 15)]      = f2bf(p0);
        plds[wave][rloc * 32 + 16 + (lane & 15)] = f2bf(p1);
        ls[i] = p0 + p1;
        mi[i] = mnew;
      }
#pragma unroll
      for (int mask = 1; mask <= 8; mask <<= 1)
#pragma unroll
        for (int i = 0; i < 8; ++i)
          ls[i] += __shfl_xor(ls[i], mask, 32);
#pragma unroll
      for (int i = 0; i < 8; ++i) li[i] = li[i] * al[i] + ls[i];
#pragma unroll
      for (int t = 0; t < 4; ++t)
#pragma unroll
        for (int i = 0; i < 8; ++i) o[t].f[i] *= al[i];

      // LDS store->load within the same wave: make ordering explicit.
      asm volatile("s_wait_dscnt 0" ::: "memory");

      // ---- P (A-frag from LDS) x Vt (B-frag from LDS) ---------------------
      Frag16 pf;
      load_frag(pf, &plds[wave][0], (lane & 15) * 32, lane);
#pragma unroll
      for (int t = 0; t < 4; ++t) {
        Frag16 vf;
        const int dcol = t * 16 + (lane & 15);
        load_frag(vf, vbuf, dcol * 32, lane);
        o[t].v = wmma_bf16(pf.v, vf.v, o[t].v);
      }
    }

    __syncthreads();  // all LDS reads done before this buffer is re-staged
  }

  // ---- epilogue: normalize, write [B,T,C] bf16 ----------------------------
  const int bidx = bh >> 4;
  const int h = bh & 15;
#pragma unroll
  for (int t = 0; t < 4; ++t)
#pragma unroll
    for (int i = 0; i < 8; ++i) {
      const int q = qt * 16 + i + ((lane >= 16) ? 8 : 0);
      const int d = t * 16 + (lane & 15);
      const float val = o[t].f[i] / li[i];
      ob[(bidx * TT + q) * DM + h * DH + d] = f2bf(val);
    }
}

// ---------------------------------------------------------------------------
// Output projection: [4096,1024]bf16 x Wt[1024,1024]bf16 + bias -> fp32 out,
// pipelined k-loop.
// ---------------------------------------------------------------------------
__global__ void __launch_bounds__(256)
proj_gemm_kernel(const unsigned short* __restrict__ ab,   // bf16 [4096,1024]
                 const unsigned short* __restrict__ wt,   // bf16 [1024,1024]
                 const float* __restrict__ bias,
                 float* __restrict__ out) {
  const int lane = threadIdx.x & 31;
  const int wave = __builtin_amdgcn_readfirstlane(threadIdx.x >> 5);
  const int NT64 = DM / 64;                   // 16
  const int tile = blockIdx.x * 8 + wave;     // 256*16 = 4096 wave-tiles
  const int tm = tile / NT64;
  const int tn64 = tile - tm * NT64;
  const int arow = tm * 16 + (lane & 15);

  F32x8 acc[4];
#pragma unroll
  for (int c = 0; c < 4; ++c)
#pragma unroll
    for (int i = 0; i < 8; ++i) acc[c].f[i] = 0.0f;

  int bcol[4];
#pragma unroll
  for (int c = 0; c < 4; ++c) bcol[c] = (tn64 * 4 + c) * 16 + (lane & 15);

  Frag16 a_cur, b_cur[4];
  load_frag(a_cur, ab, arow * DM, lane);
#pragma unroll
  for (int c = 0; c < 4; ++c) load_frag(b_cur[c], wt, bcol[c] * DM, lane);

  for (int k0 = 0; k0 < DM; k0 += 32) {
    Frag16 a_nxt, b_nxt[4];
    const int kn = k0 + 32;
    if (kn < DM) {
      load_frag(a_nxt, ab, arow * DM + kn, lane);
      __builtin_prefetch(ab + arow * DM + kn + 32, 0, 1);
#pragma unroll
      for (int c = 0; c < 4; ++c) load_frag(b_nxt[c], wt, bcol[c] * DM + kn, lane);
    }
#pragma unroll
    for (int c = 0; c < 4; ++c)
      acc[c].v = wmma_bf16(a_cur.v, b_cur[c].v, acc[c].v);
    a_cur = a_nxt;
#pragma unroll
    for (int c = 0; c < 4; ++c) b_cur[c] = b_nxt[c];
  }

#pragma unroll
  for (int c = 0; c < 4; ++c) {
    const float bias_n = bias[bcol[c]];
#pragma unroll
    for (int i = 0; i < 8; ++i) {
      const int m = tm * 16 + i + ((lane >= 16) ? 8 : 0);
      out[m * DM + bcol[c]] = acc[c].f[i] + bias_n;
    }
  }
}

// ---------------------------------------------------------------------------
// Host launcher
// ---------------------------------------------------------------------------
extern "C" void kernel_launch(void* const* d_in, const int* in_sizes, int n_in,
                              void* d_out, int out_size, void* d_ws, size_t ws_size,
                              hipStream_t stream) {
  (void)in_sizes; (void)n_in; (void)out_size; (void)ws_size;
  const float* x      = (const float*)d_in[0];
  const float* w_qkv  = (const float*)d_in[1];
  const float* b_qkv  = (const float*)d_in[2];
  const float* w_proj = (const float*)d_in[3];
  const float* b_proj = (const float*)d_in[4];
  float* out = (float*)d_out;

  char* ws = (char*)d_ws;
  unsigned short* xb     = (unsigned short*)(ws + 0);          //  8,388,608
  unsigned short* wqkvt  = (unsigned short*)(ws + 8388608);    //  6,291,456  [3072,1024]
  unsigned short* wprojt = (unsigned short*)(ws + 14680064);   //  2,097,152  [1024,1024]
  unsigned short* qb     = (unsigned short*)(ws + 16777216);   //  8,388,608  [B,H,T,Dh]
  unsigned short* kb     = (unsigned short*)(ws + 25165824);   //  8,388,608  [B,H,T,Dh]
  unsigned short* vt     = (unsigned short*)(ws + 33554432);   //  8,388,608  [B,H,Dh,T]
  unsigned short* ob     = (unsigned short*)(ws + 41943040);   //  8,388,608  [B,T,C]

  const int nx = BB * TT * DM;             // 4,194,304
  const int nwq = DM * NQKV;               // 3,145,728
  const int nwp = DM * DM;                 // 1,048,576
  f32_to_bf16_kernel<<<(nx + 255) / 256, 256, 0, stream>>>(x, xb, nx);
  f32_to_bf16_tr_kernel<<<(nwq + 255) / 256, 256, 0, stream>>>(w_qkv, wqkvt, DM, NQKV);
  f32_to_bf16_tr_kernel<<<(nwp + 255) / 256, 256, 0, stream>>>(w_proj, wprojt, DM, DM);

  // 256 x 48 wave-tiles, 8 waves/block
  qkv_gemm_kernel<<<1536, 256, 0, stream>>>(xb, wqkvt, b_qkv, qb, kb, vt);
  // 32 (b,h) x 16 blocks of 8 q-tiles = 512 blocks
  attn_kernel<<<512, 256, 0, stream>>>(qb, kb, vt, ob);
  // 256 x 16 wave-tiles, 8 waves/block
  proj_gemm_kernel<<<512, 256, 0, stream>>>(ob, wprojt, b_proj, out);
}
